// BinBlock_34024730919489
// MI455X (gfx1250) — 8-bit matrix operand order repaired and checked by reading, compiled, not run on hardware
//
#include <hip/hip_runtime.h>

typedef __attribute__((ext_vector_type(8))) int v8i;

#define C_CH   64
#define HSZ    112
#define WSZ    112
#define WPAD   114
#define BATCH  32
#define KTAPS  9
#define POS_PER_IMG (HSZ*WSZ)

#define WF_OFF     0
#define WF_BYTES   (KTAPS*4*32*32)
#define ASCALE_OFF (WF_OFF + WF_BYTES)
#define BSHIFT_OFF (ASCALE_OFF + 256)

#define LX_BYTES   (3*WPAD*C_CH)

__global__ void prep_scales(const float* __restrict__ w,
                            const float* __restrict__ bias,
                            const float* __restrict__ gamma,
                            const float* __restrict__ beta,
                            const float* __restrict__ rmean,
                            const float* __restrict__ rvar,
                            float* __restrict__ a_scale,
                            float* __restrict__ b_shift) {
    int o = threadIdx.x;
    float s = 0.f;
    for (int t = 0; t < C_CH * KTAPS; ++t) s += fabsf(w[o * (C_CH * KTAPS) + t]);
    float alpha = s * (1.0f / (C_CH * KTAPS));
    float scale = gamma[o] * rsqrtf(rvar[o] + 1e-5f);
    a_scale[o]  = alpha * scale;
    b_shift[o]  = bias[o] * scale + beta[o] - rmean[o] * scale;
}

__global__ void prep_wfrag(const float* __restrict__ w, signed char* __restrict__ bf) {
    int idx = blockIdx.x * 256 + threadIdx.x;
    if (idx >= WF_BYTES) return;
    int bb   = idx & 3;
    int v    = (idx >> 2) & 7;
    int lane = (idx >> 5) & 31;
    int nt   = (idx >> 10) & 3;
    int tap  = idx >> 12;
    int n    = nt * 16 + (lane & 15);
    int kbase = (lane >> 4) * 8;
    int K    = (v >> 1) * 16 + kbase + (v & 1) * 4 + bb;
    int kh   = tap / 3, kw = tap - 3 * (tap / 3);
    float ww = w[((n * C_CH + K) * 3 + kh) * 3 + kw];
    bf[idx] = (ww > 0.f) ? 1 : ((ww < 0.f) ? -1 : 0);
}

__global__ __launch_bounds__(224) void bingemm(const float* __restrict__ x,
                                               const signed char* __restrict__ bfrag,
                                               const float* __restrict__ a_scale,
                                               const float* __restrict__ b_shift,
                                               float* __restrict__ out) {
    __shared__ __align__(16) signed char lX[LX_BYTES];
    __shared__ __align__(32) signed char lB[WF_BYTES];

    const int tid = threadIdx.x;
    const int img = blockIdx.x / HSZ;
    const int h   = blockIdx.x % HSZ;

    __builtin_prefetch(a_scale, 0, 0);

    {
        int4  m1 = make_int4(-1, -1, -1, -1);
        int4* s  = (int4*)lX;
        for (int t = tid; t < LX_BYTES / 16; t += 224) s[t] = m1;
    }
    {
        const int4* g = (const int4*)bfrag;
        int4*       s = (int4*)lB;
        for (int t = tid; t < WF_BYTES / 16; t += 224) s[t] = g[t];
    }
    __syncthreads();

    for (int f = tid; f < 3 * C_CH * (HSZ / 4); f += 224) {
        int w4 = (f % (HSZ / 4)) << 2;
        int rc = f / (HSZ / 4);
        int c  = rc % C_CH;
        int rr = rc / C_CH;
        int hs = h - 1 + rr;
        if (hs >= 0 && hs < HSZ) {
            const float4 v = *(const float4*)(x + ((img * C_CH + c) * HSZ + hs) * WSZ + w4);
            signed char* d = lX + (rr * WPAD + w4 + 1) * C_CH + c;
            d[0 * C_CH] = (v.x > 0.f) ? 1 : ((v.x < 0.f) ? -1 : 0);
            d[1 * C_CH] = (v.y > 0.f) ? 1 : ((v.y < 0.f) ? -1 : 0);
            d[2 * C_CH] = (v.z > 0.f) ? 1 : ((v.z < 0.f) ? -1 : 0);
            d[3 * C_CH] = (v.w > 0.f) ? 1 : ((v.w < 0.f) ? -1 : 0);
        }
    }
    __syncthreads();

    const int wave = tid >> 5;
    const int lane = tid & 31;
    const int mrow = (wave << 4) + (lane & 15);
    const int koff = (lane >> 4) << 3;

    v8i acc[4] = {};
    #pragma unroll
    for (int tap = 0; tap < KTAPS; ++tap) {
        const int dr = tap / 3, dc = tap - 3 * dr;
        v8i a;
        const signed char* ab = lX + ((dr * WPAD) + mrow + dc) * C_CH + koff;
        #pragma unroll
        for (int j = 0; j < 4; ++j) {
            long long q = *(const long long*)(ab + j * 16);
            a[2 * j]     = (int)(unsigned int)(q & 0xffffffffLL);
            a[2 * j + 1] = (int)(unsigned int)((unsigned long long)q >> 32);
        }
        const v8i* bf = (const v8i*)lB;
        #pragma unroll
        for (int nt = 0; nt < 4; ++nt) {
            v8i bm = bf[(tap * 4 + nt) * 32 + lane];
            acc[nt] = __builtin_amdgcn_wmma_i32_16x16x64_iu8(
                true, a, true, bm, acc[nt], false, false);
        }
    }

    const int wout0 = (wave << 4) + ((lane >> 4) << 3);
    #pragma unroll
    for (int nt = 0; nt < 4; ++nt) {
        int   n  = nt * 16 + (lane & 15);
        float sa = a_scale[n];
        float sb = b_shift[n];
        int base = (img * C_CH + n) * POS_PER_IMG + h * WSZ + wout0;
        float4 r0 = *(const float4*)(x + base);
        float4 r1 = *(const float4*)(x + base + 4);
        float4 o0, o1;
        o0.x = (float)acc[nt][0] * sa + sb + r0.x;
        o0.y = (float)acc[nt][1] * sa + sb + r0.y;
        o0.z = (float)acc[nt][2] * sa + sb + r0.z;
        o0.w = (float)acc[nt][3] * sa + sb + r0.w;
        o1.x = (float)acc[nt][4] * sa + sb + r1.x;
        o1.y = (float)acc[nt][5] * sa + sb + r1.y;
        o1.z = (float)acc[nt][6] * sa + sb + r1.z;
        o1.w = (float)acc[nt][7] * sa + sb + r1.w;
        *(float4*)(out + base)     = o0;
        *(float4*)(out + base + 4) = o1;
    }
}

extern "C" void kernel_launch(void* const* d_in, const int* in_sizes, int n_in,
                              void* d_out, int out_size, void* d_ws, size_t ws_size,
                              hipStream_t stream) {
    const float* x     = (const float*)d_in[0];
    const float* wgt   = (const float*)d_in[1];
    const float* bias  = (const float*)d_in[2];
    const float* gamma = (const float*)d_in[3];
    const float* beta  = (const float*)d_in[4];
    const float* rmean = (const float*)d_in[5];
    const float* rvar  = (const float*)d_in[6];

    signed char* ws  = (signed char*)d_ws;
    signed char* wf  = ws + WF_OFF;
    float*       asc = (float*)(ws + ASCALE_OFF);
    float*       bsh = (float*)(ws + BSHIFT_OFF);
    float*       out = (float*)d_out;

    hipLaunchKernelGGL(prep_scales, dim3(1), dim3(64), 0, stream,
                       wgt, bias, gamma, beta, rmean, rvar, asc, bsh);
    hipLaunchKernelGGL(prep_wfrag, dim3((WF_BYTES + 255) / 256), dim3(256), 0, stream,
                       wgt, wf);
    hipLaunchKernelGGL(bingemm, dim3(BATCH * HSZ), dim3(224), 0, stream,
                       x, wf, asc, bsh, out);
}
